// Attention_15865609191618
// MI455X (gfx1250) — compile-verified, matrix operand order checked by reading, not run on hardware
//
#include <hip/hip_runtime.h>
#include <hip/hip_bf16.h>

// Problem constants (from reference)
#define NHEADS 4
#define DHD    32          // head dim
#define HID    128         // NHEADS*DHD
#define DIM    128
#define NTOK   64
#define NWIN   4096        // B * B2
#define QKV3   384         // 3*HID

typedef __attribute__((ext_vector_type(16))) __bf16 v16bf;
typedef __attribute__((ext_vector_type(8)))  float  v8f;
typedef __attribute__((ext_vector_type(4)))  unsigned int v4u;
typedef __attribute__((ext_vector_type(2)))  unsigned int v2u;
typedef __attribute__((ext_vector_type(4)))  float  v4f;
typedef unsigned int u32x4 __attribute__((ext_vector_type(4)));
typedef int          i32x4 __attribute__((ext_vector_type(4)));
typedef int          i32x8 __attribute__((ext_vector_type(8)));

#if __has_builtin(__builtin_amdgcn_tensor_load_to_lds)
#define HAVE_TDM 1
#else
#define HAVE_TDM 0
#endif

__device__ __forceinline__ unsigned short f2bf(float f) {
    unsigned int u = __float_as_uint(f);
    u += 0x7FFFu + ((u >> 16) & 1u);        // round-to-nearest-even
    return (unsigned short)(u >> 16);
}

union FragCast { v4u u[2]; v16bf v; };

// bf16 A/B fragment for v_wmma_f32_16x16x32_bf16; storage [row][k] row-major
// (B stored transposed, i.e. N-major). Per ISA 7.12.2: lane L -> row row0+L%16;
// lanes<16 hold K {0..7,16..23}, lanes>=16 hold K {8..15,24..31}:
// two contiguous 16B chunks per lane.
__device__ __forceinline__ v16bf load_frag(const unsigned short* base,
                                           int row0, int k0, int stride, int lane) {
    int r  = row0 + (lane & 15);
    int kh = (lane >> 4) << 3;             // 0 or 8
    const unsigned short* p = base + r * stride + k0 + kh;
    FragCast f;
    f.u[0] = *(const v4u*)(p);
    f.u[1] = *(const v4u*)(p + 16);
    return f.v;
}

// ---------------- weight prep: f32 -> bf16, transposed to N-major ----------------
__global__ void prep_weights(const float* __restrict__ Wqkv,
                             const float* __restrict__ Wout,
                             unsigned short* __restrict__ ws) {
    int idx = blockIdx.x * blockDim.x + threadIdx.x;
    if (idx < QKV3 * DIM) {
        int o = idx / DIM, i = idx % DIM;                  // WqkvT[o][i] = Wqkv[i][o]
        ws[o * DIM + i] = f2bf(Wqkv[i * QKV3 + o]);
    } else if (idx < QKV3 * DIM + HID * DIM) {
        int j = idx - QKV3 * DIM;
        int o = j / HID, i = j % HID;                      // WoutT[o][i] = Wout[i][o]
        ws[QKV3 * DIM + o * HID + i] = f2bf(Wout[i * DIM + o]);
    }
}

// ---------------- fused per-window attention ----------------
// LDS plan (128KB total -> 2 workgroups per WGP within the 320KB budget):
//   [  0K.. 16K) xb   64x128 bf16      | aliased by attn (0..32K) after QKV/sim
//   [ 16K.. 32K) qb   64x128 bf16      |
//   [ 32K.. 48K) kb   64x128 bf16      | aliased by obuf (32..48K) after sim
//   [ 48K.. 64K) vT   [h*32+d][tok]    | live until attn@v
//   [ 64K..128K) simf 4x64x64 f32      | aliased by x f32 TDM staging (64..96K)
__global__ __launch_bounds__(256)
void attn_kernel(const float* __restrict__ x,
                 const float* __restrict__ pos_bias,
                 const unsigned short* __restrict__ WqkvT,   // [384][128] bf16
                 const unsigned short* __restrict__ WoutT,   // [128][128] bf16
                 float* __restrict__ out) {
    extern __shared__ char smem_raw[];
    unsigned short* xb   = (unsigned short*)smem_raw;
    unsigned short* qb   = (unsigned short*)(smem_raw + 16 * 1024);
    unsigned short* kb   = (unsigned short*)(smem_raw + 32 * 1024);
    unsigned short* vT   = (unsigned short*)(smem_raw + 48 * 1024);
    float*          simf = (float*)(smem_raw + 64 * 1024);
    unsigned short* attn = (unsigned short*)smem_raw;              // alias xb+qb
    unsigned short* obuf = (unsigned short*)(smem_raw + 32 * 1024);// alias kb
    float*          xstg = (float*)(smem_raw + 64 * 1024);         // alias simf

    const int tid  = threadIdx.x;
    const int lane = tid & 31;
    const int wave = tid >> 5;                               // 8 wave32s
    const float* xg = x   + (size_t)blockIdx.x * (NTOK * DIM);
    float*       og = out + (size_t)blockIdx.x * (NTOK * DIM);

    // ---- stage x window (32KB f32) into LDS via the Tensor Data Mover ----
#if HAVE_TDM
    if (wave == 0) {
        unsigned long long ga = (unsigned long long)(const void*)xg;
        u32x4 g0;
        g0.x = 1u;                                           // count=1 (user, valid)
        g0.y = 64u * 1024u;                                  // lds_addr (dyn LDS base = 0)
        g0.z = (unsigned)(ga & 0xFFFFFFFFu);                 // global_addr[31:0]
        g0.w = (unsigned)((ga >> 32) & 0x1FFFFFFu) | (2u << 30); // [56:32] + type=2
        i32x8 g1 = {};
        g1[0] = (int)(2u << 16);                             // data_size = 4B
        g1[1] = (int)((8192u & 0xFFFFu) << 16);              // tensor_dim0[15:0]
        g1[2] = (int)((8192u >> 16) | (1u << 16));           // dim0 hi, tensor_dim1=1
        g1[3] = (int)(8192u << 16);                          // tile_dim0=8192
        g1[4] = 1;                                           // tile_dim1=1
        g1[5] = 8192;                                        // tensor_dim0_stride
        i32x4 gz = {};
#if __clang_major__ >= 23
        i32x8 gz8 = {};
        __builtin_amdgcn_tensor_load_to_lds(g0, g1, gz, gz, gz8, 0);
#else
        __builtin_amdgcn_tensor_load_to_lds(g0, g1, gz, gz, 0);
#endif
        __builtin_amdgcn_s_wait_tensorcnt(0);
    }
    __syncthreads();
#endif

    // ---- convert x to bf16 in LDS ----
#if HAVE_TDM
    const v4f* xsrc4 = (const v4f*)xstg;
#else
    const v4f* xsrc4 = (const v4f*)xg;
#endif
    #pragma unroll
    for (int i = 0; i < 8; ++i) {
        int e4 = i * 256 + tid;                              // 2048 float4s
        v4f v = xsrc4[e4];
        v2u p;
        p.x = (unsigned int)f2bf(v.x) | ((unsigned int)f2bf(v.y) << 16);
        p.y = (unsigned int)f2bf(v.z) | ((unsigned int)f2bf(v.w) << 16);
        *(v2u*)(xb + e4 * 4) = p;
    }
    __syncthreads();

    const float scale = 0.17677669529663687f;                // DH^-0.5

    // ---- QKV projection, register-blocked: each wave owns one 16-wide output
    //      column per pass; B-fragments hoisted -> each weight byte read once ----
    {   // Q pass: column nt = wave
        v16bf bw[4];
        #pragma unroll
        for (int ks = 0; ks < 4; ++ks)
            bw[ks] = load_frag(WqkvT, wave * 16, ks * 32, DIM, lane);
        #pragma unroll
        for (int mt = 0; mt < 4; ++mt) {
            v8f acc = {};
            #pragma unroll
            for (int ks = 0; ks < 4; ++ks) {
                v16bf a = load_frag(xb, mt * 16, ks * 32, DIM, lane);
                acc = __builtin_amdgcn_wmma_f32_16x16x32_bf16(false, a, false, bw[ks],
                                                              (short)0, acc, false, false);
            }
            int n  = wave * 16 + (lane & 15);
            int mb = mt * 16 + ((lane >> 4) << 3);
            #pragma unroll
            for (int r = 0; r < 8; ++r) qb[(mb + r) * DIM + n] = f2bf(acc[r] * scale);
        }
    }
    {   // K pass
        v16bf bw[4];
        #pragma unroll
        for (int ks = 0; ks < 4; ++ks)
            bw[ks] = load_frag(WqkvT + HID * DIM, wave * 16, ks * 32, DIM, lane);
        #pragma unroll
        for (int mt = 0; mt < 4; ++mt) {
            v8f acc = {};
            #pragma unroll
            for (int ks = 0; ks < 4; ++ks) {
                v16bf a = load_frag(xb, mt * 16, ks * 32, DIM, lane);
                acc = __builtin_amdgcn_wmma_f32_16x16x32_bf16(false, a, false, bw[ks],
                                                              (short)0, acc, false, false);
            }
            int n  = wave * 16 + (lane & 15);
            int mb = mt * 16 + ((lane >> 4) << 3);
            #pragma unroll
            for (int r = 0; r < 8; ++r) kb[(mb + r) * DIM + n] = f2bf(acc[r]);
        }
    }
    {   // V pass (store transposed for attn@v B-operand)
        v16bf bw[4];
        #pragma unroll
        for (int ks = 0; ks < 4; ++ks)
            bw[ks] = load_frag(WqkvT + 2 * HID * DIM, wave * 16, ks * 32, DIM, lane);
        #pragma unroll
        for (int mt = 0; mt < 4; ++mt) {
            v8f acc = {};
            #pragma unroll
            for (int ks = 0; ks < 4; ++ks) {
                v16bf a = load_frag(xb, mt * 16, ks * 32, DIM, lane);
                acc = __builtin_amdgcn_wmma_f32_16x16x32_bf16(false, a, false, bw[ks],
                                                              (short)0, acc, false, false);
            }
            int n  = wave * 16 + (lane & 15);                // n = h*32 + d
            int mb = mt * 16 + ((lane >> 4) << 3);
            #pragma unroll
            for (int r = 0; r < 8; ++r) vT[n * NTOK + (mb + r)] = f2bf(acc[r]);
        }
    }
    __syncthreads();

    // ---- sim = q @ k^T per head: 64 tiles, K=32 -> single wmma each ----
    #pragma unroll
    for (int i = 0; i < 8; ++i) {
        int t = wave + i * 8;
        int h = t >> 4, idx = t & 15, mt = idx >> 2, nt = idx & 3;
        v16bf a = load_frag(qb + h * DHD, mt * 16, 0, DIM, lane);
        v16bf b = load_frag(kb + h * DHD, nt * 16, 0, DIM, lane);
        v8f acc = {};
        acc = __builtin_amdgcn_wmma_f32_16x16x32_bf16(false, a, false, b,
                                                      (short)0, acc, false, false);
        float* sdst = simf + h * NTOK * NTOK;
        int n  = nt * 16 + (lane & 15);
        int mb = mt * 16 + ((lane >> 4) << 3);
        #pragma unroll
        for (int r = 0; r < 8; ++r) sdst[(mb + r) * NTOK + n] = acc[r];
    }
    __syncthreads();

    // ---- softmax (f32): one row per thread (4 heads * 64 rows = 256) ----
    {
        int h = tid >> 6, row = tid & 63;
        float*       srow = simf + (h * NTOK + row) * NTOK;
        const float* pb   = pos_bias + (h * NTOK + row) * NTOK;
        float mx = -3.402823466e38f;
        for (int j = 0; j < NTOK; ++j) {
            float v = srow[j] + pb[j];
            srow[j] = v;
            mx = fmaxf(mx, v);
        }
        float sum = 0.f;
        for (int j = 0; j < NTOK; ++j) {
            float e = __expf(srow[j] - mx);
            srow[j] = e;
            sum += e;
        }
        float inv = 1.0f / sum;
        unsigned short* arow = attn + (h * NTOK + row) * NTOK;
        for (int j = 0; j < NTOK; ++j) arow[j] = f2bf(srow[j] * inv);
    }
    __syncthreads();

    // ---- out_h = attn @ v: per head 4x2 tiles, K=64 (2 wmma steps) ----
    #pragma unroll
    for (int i = 0; i < 4; ++i) {
        int t = wave + i * 8;
        int h = t >> 3, idx = t & 7, mt = idx >> 1, nt = idx & 1;
        v8f acc = {};
        #pragma unroll
        for (int ks = 0; ks < 2; ++ks) {
            v16bf a = load_frag(attn + h * NTOK * NTOK, mt * 16, ks * 32, NTOK, lane);
            v16bf b = load_frag(vT   + h * DHD  * NTOK, nt * 16, ks * 32, NTOK, lane);
            acc = __builtin_amdgcn_wmma_f32_16x16x32_bf16(false, a, false, b,
                                                          (short)0, acc, false, false);
        }
        int col = h * DHD + nt * 16 + (lane & 15);
        int mb  = mt * 16 + ((lane >> 4) << 3);
        #pragma unroll
        for (int r = 0; r < 8; ++r) obuf[(mb + r) * HID + col] = f2bf(acc[r]);
    }
    __syncthreads();

    // ---- final projection, register-blocked: column nt = wave, K=128 ----
    {
        v16bf bw[4];
        #pragma unroll
        for (int ks = 0; ks < 4; ++ks)
            bw[ks] = load_frag(WoutT, wave * 16, ks * 32, HID, lane);
        #pragma unroll
        for (int mt = 0; mt < 4; ++mt) {
            v8f acc = {};
            #pragma unroll
            for (int ks = 0; ks < 4; ++ks) {
                v16bf a = load_frag(obuf, mt * 16, ks * 32, HID, lane);
                acc = __builtin_amdgcn_wmma_f32_16x16x32_bf16(false, a, false, bw[ks],
                                                              (short)0, acc, false, false);
            }
            int n  = wave * 16 + (lane & 15);
            int mb = mt * 16 + ((lane >> 4) << 3);
            #pragma unroll
            for (int r = 0; r < 8; ++r) og[(mb + r) * DIM + n] = acc[r];
        }
    }
}

extern "C" void kernel_launch(void* const* d_in, const int* in_sizes, int n_in,
                              void* d_out, int out_size, void* d_ws, size_t ws_size,
                              hipStream_t stream) {
    const float* x        = (const float*)d_in[0];
    const float* pos_bias = (const float*)d_in[1];
    const float* Wqkv     = (const float*)d_in[2];   // (128, 384) row-major
    const float* Wout     = (const float*)d_in[3];   // (128, 128) row-major
    unsigned short* wbf   = (unsigned short*)d_ws;   // 96KB WqkvT + 32KB WoutT

    int nprep = QKV3 * DIM + HID * DIM;              // 65536
    prep_weights<<<(nprep + 255) / 256, 256, 0, stream>>>(Wqkv, Wout, wbf);

    size_t lds_bytes = 128 * 1024;                   // 2 workgroups per WGP (320KB pool)
    attn_kernel<<<NWIN, 256, lds_bytes, stream>>>(x, pos_bias,
                                                  wbf, wbf + QKV3 * DIM,
                                                  (float*)d_out);
}
